// MultiHeadAttention_3100966388128
// MI455X (gfx1250) — matrix operand order repaired and checked by reading, not compiled, not run on hardware
//
#include <hip/hip_runtime.h>
#include <hip/hip_bf16.h>

// MI455X (gfx1250) MHA block: bf16 WMMA GEMMs + flash-attention + fused proj/LN.
// Compute-bound (~43 GFLOP, ~55MB working set << 192MB L2): all matmul FLOPs go
// through v_wmma_f32_16x16x32_bf16 (fp32 accum). Attention K/V chunks are staged
// cooperatively into LDS via GLOBAL_LOAD_ASYNC_TO_LDS_B128 (ASYNCcnt), which the
// toolchain exposes as a builtin taking (v4i AS1*, v4i AS3*, imm, imm).

typedef __attribute__((ext_vector_type(16))) __bf16 bf16x16;
typedef __attribute__((ext_vector_type(8)))  __bf16 bf16x8;
typedef __attribute__((ext_vector_type(8)))  float  f32x8;
typedef __attribute__((ext_vector_type(4)))  float  f32x4;

typedef int v4i __attribute__((vector_size(16)));
typedef __attribute__((address_space(1))) v4i* gptr_v4i;
typedef __attribute__((address_space(3))) v4i* lptr_v4i;

#define DIM_   256
#define NTOK   2048
#define INNER  2048   // DIM * NUM_HEADS
#define NH     8
#define HD     256    // per-head width (per source reshape)
#define SCALE_ 0.17677669529663687f   // 32^-0.5 (HEAD_DIM=DIM/NUM_HEADS=32 per source)
#define LN_EPS 1e-5f

union AB16 { bf16x16 v; bf16x8 h[2]; };

__device__ __forceinline__ f32x8 zero8() {
  f32x8 z;
#pragma unroll
  for (int i = 0; i < 8; ++i) z[i] = 0.0f;
  return z;
}

__device__ __forceinline__ f32x8 wmma_bf16(bf16x16 a, bf16x16 b, f32x8 c) {
  // (neg_a, A, neg_b, B, c_mod, C, reuse_a, reuse_b)
  return __builtin_amdgcn_wmma_f32_16x16x32_bf16(false, a, false, b, (short)0, c,
                                                 false, false);
}

// 16-byte global -> LDS copy. Async (ASYNCcnt) path on gfx1250 toolchains that
// expose the builtin; otherwise plain load + ds_store (ordered by the fence in
// __syncthreads()).
__device__ __forceinline__ void cp16_g2l(__bf16* lds_dst, const __bf16* gsrc) {
#if defined(__gfx1250__) && __has_builtin(__builtin_amdgcn_global_load_async_to_lds_b128)
  __builtin_amdgcn_global_load_async_to_lds_b128(
      (gptr_v4i)(const __attribute__((address_space(1))) void*)gsrc,
      (lptr_v4i)(__attribute__((address_space(3))) void*)lds_dst,
      /*offset=*/0, /*cpol=*/0);
#else
  *(bf16x8*)lds_dst = *(const bf16x8*)gsrc;
#endif
}

__device__ __forceinline__ void wait_async0() {
#if defined(__gfx1250__) && __has_builtin(__builtin_amdgcn_global_load_async_to_lds_b128)
#if __has_builtin(__builtin_amdgcn_s_wait_asynccnt)
  __builtin_amdgcn_s_wait_asynccnt(0);
#else
  asm volatile("s_wait_asynccnt 0" ::: "memory");
#endif
#endif
}

// ---------------------------------------------------------------- convert ----
__global__ void k_f32_to_bf16(const float* __restrict__ src,
                              __bf16* __restrict__ dst, int n) {
  int i = blockIdx.x * blockDim.x + threadIdx.x;
  int stride = gridDim.x * blockDim.x;
  for (; i < n; i += stride) dst[i] = (__bf16)src[i];
}

// -------------------------------------------------------------- QKV GEMM ----
// C[2048 x 2048] = A[2048 x 256] @ W[2048 x 256]^T + bias.
// transpose_out==0 : out[m*INNER + n] (row-major, for Q/K)
// transpose_out==1 : out[n*NTOK  + m] (dim-major,  for V -> contiguous PV B-frags)
__global__ void __launch_bounds__(128)
k_gemm_qkv(const __bf16* __restrict__ A, const __bf16* __restrict__ W,
           const float* __restrict__ bias, __bf16* __restrict__ out,
           int transpose_out) {
  const int lane = threadIdx.x & 31;
  const int wave = threadIdx.x >> 5;
  const int m0   = blockIdx.x * 16;
  const int n0   = blockIdx.y * 256 + wave * 64;
  const int mrow = lane & 15, hk = lane >> 4;

  f32x8 acc[4];
#pragma unroll
  for (int t = 0; t < 4; ++t) acc[t] = zero8();

  const __bf16* arow = A + (size_t)(m0 + mrow) * DIM_;
#pragma unroll
  for (int kk = 0; kk < 8; ++kk) {          // K = 256 = 8 * 32
    AB16 a;
    a.h[0] = *(const bf16x8*)(arow + kk * 32 + hk * 16);
    a.h[1] = *(const bf16x8*)(arow + kk * 32 + hk * 16 + 8);
#pragma unroll
    for (int t = 0; t < 4; ++t) {
      const int n = n0 + t * 16 + mrow;
      bf16x16 b = *(const bf16x16*)(W + (size_t)n * DIM_ + kk * 32 + hk * 16);
      acc[t] = wmma_bf16(a.v, b, acc[t]);
    }
  }

#pragma unroll
  for (int t = 0; t < 4; ++t) {
    const int n = n0 + t * 16 + mrow;
    const float bn = bias[n];
    if (!transpose_out) {
#pragma unroll
      for (int r = 0; r < 8; ++r) {
        const int m = m0 + r + hk * 8;
        out[(size_t)m * INNER + n] = (__bf16)(acc[t][r] + bn);
      }
    } else {
      bf16x8 v;
#pragma unroll
      for (int r = 0; r < 8; ++r) v[r] = (__bf16)(acc[t][r] + bn);
      *(bf16x8*)(out + (size_t)n * NTOK + m0 + hk * 8) = v;
    }
  }
}

// ------------------------------------------------------------- attention ----
// Block = 4 waves (one 16-query block each) x same head. Flash-style over 64
// chunks of 32 keys. K/V chunk staged once per block into LDS (async-to-LDS).
__global__ void __launch_bounds__(128)
k_attn(const __bf16* __restrict__ Q, const __bf16* __restrict__ K,
       const __bf16* __restrict__ Vt, __bf16* __restrict__ X) {
  __shared__ __align__(32) __bf16 Ks[32 * 264];     // K chunk: 32 keys x 256 dims
  __shared__ __align__(32) __bf16 Vs[256 * 40];     // V chunk: 256 dims x 32 keys
  __shared__ __align__(16) float  Sw[4][16 * 34];   // per-wave fp32 S tile 16x32
  __shared__ __align__(16) __bf16 Pb[4][16 * 40];   // per-wave bf16 P tile 16x32
  __shared__ __align__(16) float  Mst[4][16], Lst[4][16], Alp[4][16];

  const int tid  = threadIdx.x;
  const int lane = tid & 31;
  const int w    = tid >> 5;
  const int qb   = blockIdx.x * 4 + w;
  const int h    = blockIdx.y;
  const int mrow = lane & 15, hk = lane >> 4;

  if (lane < 16) { Mst[w][lane] = -1e30f; Lst[w][lane] = 0.0f; }

  f32x8 o[16];
#pragma unroll
  for (int t = 0; t < 16; ++t) o[t] = zero8();

  const __bf16* qrow = Q + (size_t)(qb * 16 + mrow) * INNER + h * HD;

  for (int kc = 0; kc < 64; ++kc) {
    // ---- cooperative stage: K chunk (32x256) + V chunk (256x32) -> LDS ----
#pragma unroll
    for (int j = 0; j < 8; ++j) {
      const int idx = j * 128 + tid;               // 0..1023 16B-transfers each
      const int kr = idx >> 5, ko = idx & 31;      // K: 32 rows x 32 x 16B
      cp16_g2l(&Ks[kr * 264 + ko * 8],
               K + (size_t)(kc * 32 + kr) * INNER + h * HD + ko * 8);
      const int vr = idx >> 2, vo = idx & 3;       // V: 256 rows x 4 x 16B
      cp16_g2l(&Vs[vr * 40 + vo * 8],
               Vt + (size_t)(h * HD + vr) * NTOK + kc * 32 + vo * 8);
    }
    wait_async0();
    __syncthreads();

    // ---- S(16x32) = Qblk @ Kchunk^T ----
    f32x8 s0 = zero8(), s1 = zero8();
#pragma unroll
    for (int kk = 0; kk < 8; ++kk) {
      AB16 a, b0, b1;
      a.h[0] = *(const bf16x8*)(qrow + kk * 32 + hk * 16);
      a.h[1] = *(const bf16x8*)(qrow + kk * 32 + hk * 16 + 8);
      const __bf16* kr0 = &Ks[mrow * 264 + kk * 32 + hk * 16];
      const __bf16* kr1 = &Ks[(16 + mrow) * 264 + kk * 32 + hk * 16];
      b0.h[0] = *(const bf16x8*)kr0;  b0.h[1] = *(const bf16x8*)(kr0 + 8);
      b1.h[0] = *(const bf16x8*)kr1;  b1.h[1] = *(const bf16x8*)(kr1 + 8);
      s0 = wmma_bf16(a.v, b0.v, s0);
      s1 = wmma_bf16(a.v, b1.v, s1);
    }

    // ---- spill scaled S to LDS (C layout: lane=col, vgpr r -> row r+8*hk) ----
    float* sw = &Sw[w][0];
#pragma unroll
    for (int r = 0; r < 8; ++r) {
      const int m = r + hk * 8;
      sw[m * 34 + mrow]      = s0[r] * SCALE_;
      sw[m * 34 + 16 + mrow] = s1[r] * SCALE_;
    }
    asm volatile("s_wait_dscnt 0" ::: "memory");   // wave-local LDS ordering

    // ---- online softmax (lanes 0..15 own one row each) ----
    if (lane < 16) {
      const int m = lane;
      float rmax = -1e30f;
#pragma unroll
      for (int c = 0; c < 32; ++c) rmax = fmaxf(rmax, sw[m * 34 + c]);
      const float mold = Mst[w][m];
      const float mnew = fmaxf(mold, rmax);
      float ls = 0.0f;
#pragma unroll
      for (int c = 0; c < 32; ++c) {
        const float p = __expf(sw[m * 34 + c] - mnew);
        Pb[w][m * 40 + c] = (__bf16)p;
        ls += p;
      }
      const float al = __expf(mold - mnew);
      Mst[w][m] = mnew;
      Lst[w][m] = Lst[w][m] * al + ls;
      Alp[w][m] = al;
    }
    asm volatile("s_wait_dscnt 0" ::: "memory");

    // ---- rescale O and accumulate O += P @ Vchunk ----
    f32x4 a0 = *(const f32x4*)&Alp[w][hk * 8];
    f32x4 a1 = *(const f32x4*)&Alp[w][hk * 8 + 4];
    AB16 p;
    p.h[0] = *(const bf16x8*)&Pb[w][mrow * 40 + hk * 16];
    p.h[1] = *(const bf16x8*)&Pb[w][mrow * 40 + hk * 16 + 8];
#pragma unroll
    for (int t = 0; t < 16; ++t) {
#pragma unroll
      for (int r = 0; r < 4; ++r) { o[t][r] *= a0[r]; o[t][r + 4] *= a1[r]; }
      const __bf16* vr = &Vs[(t * 16 + mrow) * 40 + hk * 16];
      AB16 b;
      b.h[0] = *(const bf16x8*)vr;  b.h[1] = *(const bf16x8*)(vr + 8);
      o[t] = wmma_bf16(p.v, b.v, o[t]);
    }
    __syncthreads();   // all waves done reading Ks/Vs before next-chunk stage
  }

  // ---- normalize by running sum and store bf16 attention output ----
  if (lane < 16) Alp[w][lane] = 1.0f / Lst[w][lane];
  asm volatile("s_wait_dscnt 0" ::: "memory");
  f32x4 i0 = *(const f32x4*)&Alp[w][hk * 8];
  f32x4 i1 = *(const f32x4*)&Alp[w][hk * 8 + 4];
#pragma unroll
  for (int t = 0; t < 16; ++t) {
#pragma unroll
    for (int r = 0; r < 8; ++r) {
      const float inv = (r < 4) ? i0[r] : i1[r - 4];
      const int m = qb * 16 + r + hk * 8;
      X[(size_t)m * INNER + h * HD + t * 16 + mrow] = (__bf16)(o[t][r] * inv);
    }
  }
}

// --------------------------------------- projection + ReLU + residual + LN ----
__global__ void __launch_bounds__(128)
k_proj_ln(const __bf16* __restrict__ X, const __bf16* __restrict__ Wp,
          const float* __restrict__ bias, const float* __restrict__ resid,
          const float* __restrict__ g, const float* __restrict__ beta,
          float* __restrict__ out) {
  __shared__ __align__(16) float Xs[16 * 260];
  __shared__ float Mu[16], Rs[16];

  const int lane = threadIdx.x & 31;
  const int w    = threadIdx.x >> 5;
  const int m0   = blockIdx.x * 16;
  const int n0   = w * 64;
  const int mrow = lane & 15, hk = lane >> 4;

  f32x8 acc[4];
#pragma unroll
  for (int t = 0; t < 4; ++t) acc[t] = zero8();

  const __bf16* arow = X + (size_t)(m0 + mrow) * INNER;
  for (int kk = 0; kk < 64; ++kk) {         // K = 2048 = 64 * 32
    __builtin_prefetch(arow + (kk + 1) * 32, 0, 1);
    AB16 a;
    a.h[0] = *(const bf16x8*)(arow + kk * 32 + hk * 16);
    a.h[1] = *(const bf16x8*)(arow + kk * 32 + hk * 16 + 8);
#pragma unroll
    for (int t = 0; t < 4; ++t) {
      const int n = n0 + t * 16 + mrow;
      bf16x16 b = *(const bf16x16*)(Wp + (size_t)n * INNER + kk * 32 + hk * 16);
      acc[t] = wmma_bf16(a.v, b, acc[t]);
    }
  }

#pragma unroll
  for (int t = 0; t < 4; ++t) {
    const int n = n0 + t * 16 + mrow;
    const float bn = bias[n];
#pragma unroll
    for (int r = 0; r < 8; ++r) {
      const int m = r + hk * 8;
      float v = acc[t][r] + bn;
      v = v > 0.0f ? v : 0.0f;                           // ReLU
      v += resid[(size_t)(m0 + m) * DIM_ + n];           // residual
      Xs[m * 260 + n] = v;
    }
  }
  __syncthreads();

  if (threadIdx.x < 16) {
    const int m = threadIdx.x;
    float s = 0.0f, ss = 0.0f;
    for (int c = 0; c < DIM_; ++c) { const float v = Xs[m * 260 + c]; s += v; ss += v * v; }
    const float mu  = s * (1.0f / DIM_);
    const float var = ss * (1.0f / DIM_) - mu * mu;
    Mu[m] = mu;
    Rs[m] = rsqrtf(var + LN_EPS);
  }
  __syncthreads();

  for (int i = threadIdx.x; i < 16 * DIM_; i += blockDim.x) {
    const int m = i >> 8, c = i & 255;
    out[(size_t)(m0 + m) * DIM_ + c] =
        (Xs[m * 260 + c] - Mu[m]) * Rs[m] * g[c] + beta[c];
  }
}

// ------------------------------------------------------------------ launch ----
extern "C" void kernel_launch(void* const* d_in, const int* in_sizes, int n_in,
                              void* d_out, int out_size, void* d_ws, size_t ws_size,
                              hipStream_t stream) {
  (void)in_sizes; (void)n_in; (void)out_size; (void)ws_size;
  const float* curr = (const float*)d_in[0];
  const float* reff = (const float*)d_in[1];
  const float* Wq   = (const float*)d_in[2];
  const float* Wqb  = (const float*)d_in[3];
  const float* Wk   = (const float*)d_in[4];
  const float* Wkb  = (const float*)d_in[5];
  const float* Wv   = (const float*)d_in[6];
  const float* Wvb  = (const float*)d_in[7];
  const float* Pw   = (const float*)d_in[8];
  const float* Pbv  = (const float*)d_in[9];
  const float* lng  = (const float*)d_in[10];
  const float* lnb  = (const float*)d_in[11];
  float* out = (float*)d_out;

  char* ws = (char*)d_ws;
  size_t off = 0;
  auto alloc = [&](size_t bytes) {
    void* p = ws + off;
    off += (bytes + 255) & ~(size_t)255;
    return p;
  };
  const size_t SMALL = (size_t)NTOK * DIM_ * 2;    // 1 MB (2048x256 bf16)
  const size_t BIG   = (size_t)NTOK * INNER * 2;   // 8 MB (2048x2048 bf16)
  __bf16* cA  = (__bf16*)alloc(SMALL);   // curr_feat bf16
  __bf16* rA  = (__bf16*)alloc(SMALL);   // ref_feat  bf16
  __bf16* wqb = (__bf16*)alloc(SMALL);
  __bf16* wkb = (__bf16*)alloc(SMALL);
  __bf16* wvb = (__bf16*)alloc(SMALL);
  __bf16* pwb = (__bf16*)alloc(SMALL);   // proj_w (256x2048) bf16
  __bf16* Qm  = (__bf16*)alloc(BIG);     // Q  (tok x inner), row-major
  __bf16* Km  = (__bf16*)alloc(BIG);     // K  (tok x inner), row-major
  __bf16* Vt  = (__bf16*)alloc(BIG);     // V^T (inner x tok)
  __bf16* Xm  = (__bf16*)alloc(BIG);     // attention output (tok x inner)

  const int NCVT = NTOK * DIM_;  // 524288 (same count for every converted buffer)
  k_f32_to_bf16<<<512, 256, 0, stream>>>(curr, cA,  NCVT);
  k_f32_to_bf16<<<512, 256, 0, stream>>>(reff, rA,  NCVT);
  k_f32_to_bf16<<<512, 256, 0, stream>>>(Wq,   wqb, NCVT);
  k_f32_to_bf16<<<512, 256, 0, stream>>>(Wk,   wkb, NCVT);
  k_f32_to_bf16<<<512, 256, 0, stream>>>(Wv,   wvb, NCVT);
  k_f32_to_bf16<<<512, 256, 0, stream>>>(Pw,   pwb, NCVT);

  dim3 gq(128, 8);
  k_gemm_qkv<<<gq, 128, 0, stream>>>(cA, wqb, Wqb, Qm, 0);
  k_gemm_qkv<<<gq, 128, 0, stream>>>(rA, wkb, Wkb, Km, 0);
  k_gemm_qkv<<<gq, 128, 0, stream>>>(rA, wvb, Wvb, Vt, 1);   // transposed store

  dim3 ga(32, NH);  // 4 query-blocks per block (1 per wave) x 8 heads
  k_attn<<<ga, 128, 0, stream>>>(Qm, Km, Vt, Xm);

  k_proj_ln<<<128, 128, 0, stream>>>(Xm, pwb, Pbv, curr, lng, lnb, out);
}
